// DSTA_36558761624545
// MI455X (gfx1250) — matrix operand order repaired and checked by reading, not compiled, not run on hardware
//
#include <hip/hip_runtime.h>
#include <hip/hip_bf16.h>
#include <math.h>

#define B_   10
#define T_   100
#define NB_  20
#define D_   4096
#define H_   512
#define WIN_ 10
#define NROWS (B_*T_*NB_)        /* 20000 */
#define MTILE 128
#define NTILE 128
#define KC    64
#define MTILES ((NROWS + MTILE - 1) / MTILE)   /* 157 */
#define NROWS_PAD (MTILES * MTILE)             /* 20096 */

typedef __bf16 bf16;
typedef __attribute__((ext_vector_type(16))) __bf16 bf16x16;
typedef __attribute__((ext_vector_type(8)))  __bf16 bf16x8;
typedef __attribute__((ext_vector_type(8)))  float  fx8;
typedef unsigned int v4u __attribute__((ext_vector_type(4)));
typedef int          v8i __attribute__((ext_vector_type(8)));
typedef int          v4i __attribute__((ext_vector_type(4)));

#if __has_builtin(__builtin_amdgcn_tensor_load_to_lds)
#define TDM_OK 1
#else
#define TDM_OK 0
#endif

// ---------------- WMMA fragment loaders (CDNA5 16x16x32 bf16 layouts) ----------------
__device__ __forceinline__ bf16x16 frag_a16x32(const bf16* tile, int ld) {
  const int lane = threadIdx.x & 31;
  const bf16* q = tile + (size_t)(lane & 15) * ld + (lane >> 4) * 8;
  bf16x8 lo = *(const bf16x8*)(q);
  bf16x8 hi = *(const bf16x8*)(q + 16);
  bf16x16 f;
#pragma unroll
  for (int i = 0; i < 8; ++i) { f[i] = lo[i]; f[i + 8] = hi[i]; }
  return f;
}
__device__ __forceinline__ bf16x16 frag_b32x16(const bf16* tileT, int ld) {
  const int lane = threadIdx.x & 31;
  const bf16* q = tileT + (size_t)(lane & 15) * ld + (lane >> 4) * 8;
  bf16x8 lo = *(const bf16x8*)(q);
  bf16x8 hi = *(const bf16x8*)(q + 16);
  bf16x16 f;
#pragma unroll
  for (int i = 0; i < 8; ++i) { f[i] = lo[i]; f[i + 8] = hi[i]; }
  return f;
}

// One 16x16 output tile: A (LDS/generic bf16, lda) x BT tile (bf16, ldb), K reduce
__device__ __forceinline__ fx8 gemm_tile(const bf16* Arows, int lda,
                                         const bf16* BTtile, int ldb, int K) {
  fx8 acc;
#pragma unroll
  for (int r = 0; r < 8; ++r) acc[r] = 0.f;
  for (int ks = 0; ks < K; ks += 32) {
    bf16x16 a = frag_a16x32(Arows + ks, lda);
    bf16x16 b = frag_b32x16(BTtile + ks, ldb);
    acc = __builtin_amdgcn_wmma_f32_16x16x32_bf16(false, a, false, b,
                                                  (short)0, acc, false, false);
  }
  return acc;
}
__device__ __forceinline__ void store_acc_f32(fx8 a, float* C, int ldc, int row0, int col0) {
  const int lane = threadIdx.x & 31;
  const int col = col0 + (lane & 15);
  const int rbase = row0 + (lane >> 4) * 8;
#pragma unroll
  for (int r = 0; r < 8; ++r) C[(rbase + r) * ldc + col] = a[r];
}
__device__ __forceinline__ void store_acc_bf16(fx8 a, bf16* C, int ldc, int row0, int col0) {
  const int lane = threadIdx.x & 31;
  const int col = col0 + (lane & 15);
  const int rbase = row0 + (lane >> 4) * 8;
#pragma unroll
  for (int r = 0; r < 8; ++r) C[(rbase + r) * ldc + col] = (bf16)a[r];
}

// ---------------- TDM: async 2-D tile (rows x kc bf16) global -> LDS -----------------
#if TDM_OK
__device__ __forceinline__ void tdm_load_2d(unsigned lds_off, const bf16* gsrc,
                                            unsigned tile_k, unsigned tile_rows,
                                            unsigned row_stride_elems)
{
  unsigned long long ga = (unsigned long long)(const void*)gsrc;
  v4u g0;
  g0[0] = 1u;
  g0[1] = lds_off;
  g0[2] = (unsigned)(ga & 0xffffffffu);
  g0[3] = (unsigned)((ga >> 32) & 0x1ffffffu) | (2u << 30);
  v8i g1;
  g1[0] = (int)(1u << 16);                                         // data_size = 2B
  g1[1] = (int)((tile_k & 0xffffu) << 16);                         // tensor_dim0 lo
  g1[2] = (int)((tile_k >> 16) | ((tile_rows & 0xffffu) << 16));   // td0 hi | td1 lo
  g1[3] = (int)((tile_rows >> 16) | (tile_k << 16));               // td1 hi | tile_dim0
  g1[4] = (int)tile_rows;                                          // tile_dim1
  g1[5] = (int)row_stride_elems;                                   // dim0 stride lo32
  g1[6] = 0;
  g1[7] = 0;
  v4i z4 = {0, 0, 0, 0};
  v8i z8 = {0, 0, 0, 0, 0, 0, 0, 0};
  __builtin_amdgcn_tensor_load_to_lds(g0, g1, z4, z4, z8, 0);
}
#endif

// ------ GEMM: C[M x 512] = act(A[Mpad x K](bf16) * BT[512 x K]^T + bias) -------------
__global__ void __launch_bounds__(256) wmma_gemm_n512(
    const bf16* __restrict__ A, const bf16* __restrict__ BT,
    const float* __restrict__ bias, float* __restrict__ Cf,
    bf16* __restrict__ Cbf, int K, int relu, int M)
{
  __shared__ bf16 sA[2][MTILE * KC];
  __shared__ bf16 sB[2][NTILE * KC];

  const int mt   = blockIdx.x >> 2;
  const int nt   = blockIdx.x & 3;
  const int m0   = mt * MTILE;
  const int n0   = nt * NTILE;
  const int wv   = threadIdx.x >> 5;
  const int lane = threadIdx.x & 31;
  const int wm   = wv & 3;
  const int wn   = wv >> 2;

  fx8 acc[2][4];
#pragma unroll
  for (int m = 0; m < 2; ++m)
#pragma unroll
    for (int n = 0; n < 4; ++n)
#pragma unroll
      for (int r = 0; r < 8; ++r) acc[m][n][r] = 0.f;

  const int nk = K / KC;

#if TDM_OK
  if (wv == 0) {
    tdm_load_2d((unsigned)(unsigned long long)(void*)&sA[0][0],
                A + (size_t)m0 * K, KC, MTILE, (unsigned)K);
    tdm_load_2d((unsigned)(unsigned long long)(void*)&sB[0][0],
                BT + (size_t)n0 * K, KC, NTILE, (unsigned)K);
  }
#endif

  for (int kk = 0; kk < nk; ++kk) {
#if TDM_OK
    const int cur = kk & 1, nxt = cur ^ 1;
    if (wv == 0) {
      if (kk + 1 < nk) {
        tdm_load_2d((unsigned)(unsigned long long)(void*)&sA[nxt][0],
                    A + (size_t)m0 * K + (kk + 1) * KC, KC, MTILE, (unsigned)K);
        tdm_load_2d((unsigned)(unsigned long long)(void*)&sB[nxt][0],
                    BT + (size_t)n0 * K + (kk + 1) * KC, KC, NTILE, (unsigned)K);
        __builtin_amdgcn_s_wait_tensorcnt(2);
      } else {
        __builtin_amdgcn_s_wait_tensorcnt(0);
      }
    }
    __syncthreads();
#else
    const int cur = 0;
    __syncthreads();
    for (int i = threadIdx.x; i < MTILE * KC / 8; i += 256)
      ((bf16x8*)&sA[0][0])[i] = ((const bf16x8*)(A + (size_t)m0 * K))[ (i >> 3) * (K >> 3) + (size_t)(kk * KC / 8) + (i & 7) ];
    for (int i = threadIdx.x; i < NTILE * KC / 8; i += 256)
      ((bf16x8*)&sB[0][0])[i] = ((const bf16x8*)(BT + (size_t)n0 * K))[ (i >> 3) * (K >> 3) + (size_t)(kk * KC / 8) + (i & 7) ];
    __syncthreads();
#endif

    const bf16* la = &sA[cur][0];
    const bf16* lb = &sB[cur][0];
#pragma unroll
    for (int k2 = 0; k2 < KC; k2 += 32) {
      bf16x16 af0 = frag_a16x32(la + (wm * 32 +  0) * KC + k2, KC);
      bf16x16 af1 = frag_a16x32(la + (wm * 32 + 16) * KC + k2, KC);
#pragma unroll
      for (int nn = 0; nn < 4; ++nn) {
        bf16x16 bfr = frag_b32x16(lb + (wn * 64 + nn * 16) * KC + k2, KC);
        acc[0][nn] = __builtin_amdgcn_wmma_f32_16x16x32_bf16(false, af0, false, bfr,
                                                             (short)0, acc[0][nn], false, false);
        acc[1][nn] = __builtin_amdgcn_wmma_f32_16x16x32_bf16(false, af1, false, bfr,
                                                             (short)0, acc[1][nn], false, false);
      }
    }
    __syncthreads();
  }

  const int ccol = lane & 15;
  const int mh   = lane >> 4;
#pragma unroll
  for (int m = 0; m < 2; ++m) {
#pragma unroll
    for (int nn = 0; nn < 4; ++nn) {
      int n = n0 + wn * 64 + nn * 16 + ccol;
      float bv = bias[n];
#pragma unroll
      for (int r = 0; r < 8; ++r) {
        int row = m0 + wm * 32 + m * 16 + mh * 8 + r;
        if (row < M) {
          float v = acc[m][nn][r] + bv;
          if (relu) v = fmaxf(v, 0.f);
          size_t o = (size_t)row * 512 + n;
          if (Cf)  Cf[o]  = v;
          if (Cbf) Cbf[o] = (bf16)v;
        }
      }
    }
  }
}

// ------------- x -> bf16 conversion fused with zo-flag (row-sum != 0) ----------------
__global__ void convert_zo(const float* __restrict__ X, bf16* __restrict__ Xbf,
                           float* __restrict__ flags)
{
  __shared__ float red[256];
  const int row = blockIdx.x;
  const float* xr = X + (size_t)row * D_;
  bf16* br = Xbf + (size_t)row * D_;
  float s = 0.f;
  for (int i = threadIdx.x; i < D_; i += 256) { float v = xr[i]; br[i] = (bf16)v; s += v; }
  red[threadIdx.x] = s;
  __syncthreads();
  for (int st = 128; st > 0; st >>= 1) {
    if (threadIdx.x < st) red[threadIdx.x] += red[threadIdx.x + st];
    __syncthreads();
  }
  if (threadIdx.x == 0) flags[row] = (red[0] != 0.f) ? 1.f : 0.f;
}

// W (K x N, f32, row-major) -> WT (N x K, bf16)
__global__ void transpose_bf16(const float* __restrict__ W, bf16* __restrict__ WT, int K, int N)
{
  int idx = blockIdx.x * 256 + threadIdx.x;
  if (idx < K * N) { int n = idx / K; int k = idx - n * K; WT[idx] = (bf16)W[(size_t)k * N + n]; }
}

// wcatT[o][c] = c<1024 ? wih[o][c] : whh[o][c-1024]   (1536 x 1536 bf16)
__global__ void build_wcat(const float* __restrict__ wih, const float* __restrict__ whh,
                           bf16* __restrict__ wcatT)
{
  int idx = blockIdx.x * 256 + threadIdx.x;
  if (idx < 1536 * 1536) {
    int o = idx / 1536, c = idx - o * 1536;
    float v = (c < 1024) ? wih[(size_t)o * 1024 + c] : whh[(size_t)o * 512 + (c - 1024)];
    wcatT[idx] = (bf16)v;
  }
}

__global__ void cast_bf16(const float* __restrict__ src, bf16* __restrict__ dst, int n)
{
  int idx = blockIdx.x * 256 + threadIdx.x;
  if (idx < n) dst[idx] = (bf16)src[idx];
}

// ---------------- Sequential scan: 16 waves, WMMA for all per-step GEMMs -------------
// LDS layout (bytes):
//   0      sh_h   f32 10x512      (20480)
//   20480  sh_al  f32 192         (768)
//   21248  sh_t64 f32 640         (2560)
//   23808  sh_o   f32 32          (128)
//   23936  sh_ce  f32 16          (64)
//   24000  scratch (65536): hw f32 16x512 | lg f32 112x128 | gsum bf16 16x1536 + hg bf16 16x512
//   89536  abuf   bf16 16x1536    (49152)   [xin | h] rows 10..15 zero
//   138688 tbuf   bf16 112x512    (114688)  tanh ring (100 rows) + 12 zero pad rows
#define SCAN_SMEM 253376
__global__ void __launch_bounds__(512) scan_kernel(
    const bf16*  __restrict__ XEbf, const float* __restrict__ IP,
    const float* __restrict__ flags,
    const bf16*  __restrict__ awaT, const float* __restrict__ att_w,
    const bf16*  __restrict__ fwT,  const bf16* __restrict__ wcatT,
    const bf16*  __restrict__ whhg,
    const float* __restrict__ bih, const float* __restrict__ bhh,
    const float* __restrict__ d1w, const float* __restrict__ d1b,
    const float* __restrict__ d2w, const float* __restrict__ d2b,
    const float* __restrict__ y,  const float* __restrict__ toa,
    float* __restrict__ bufG,     // ws: 10 slots x 10 x 512 f32 ring of h history
    float* __restrict__ out)      // out[0]=ce, out[1]=aux, out+2 outs, out+2002 hids
{
  extern __shared__ unsigned char smem[];
  float* sh_h   = (float*)smem;
  float* sh_al  = (float*)(smem + 20480);
  float* sh_t64 = (float*)(smem + 21248);
  float* sh_o   = (float*)(smem + 23808);
  float* sh_ce  = (float*)(smem + 23936);
  unsigned char* scr = smem + 24000;
  float* sc_hw   = (float*)scr;            // 16x512 f32
  float* sc_lg   = (float*)scr;            // 112x128 f32
  bf16*  sc_gsum = (bf16*)scr;             // 16x1536
  bf16*  sc_hg   = (bf16*)(scr + 49152);   // 16x512
  bf16*  abuf = (bf16*)(smem + 89536);     // 16x1536
  bf16*  tbuf = (bf16*)(smem + 138688);    // 112x512

  const int tid = threadIdx.x;
  const int wv  = tid >> 5;
  const int NT  = 512;

  for (int i = tid; i < B_*H_; i += NT) sh_h[i] = 0.f;
  for (int i = tid; i < 16*1536; i += NT) abuf[i] = (bf16)0.f;
  for (int i = tid; i < 112*512; i += NT) tbuf[i] = (bf16)0.f;
  if (tid == 0) sh_ce[0] = 0.f;
  __syncthreads();

  for (int t = 0; t < T_; ++t) {
    const int slot_new = t % WIN_;
    // ---- A: hw = h @ att_wa  (WMMA, A = abuf h-section) ----
    for (int nt = wv; nt < 32; nt += 16) {
      fx8 acc = gemm_tile(abuf + 1024, 1536, awaT + (size_t)nt * 16 * 512, 512, 512);
      store_acc_f32(acc, sc_hw, 512, 0, nt * 16);
    }
    __syncthreads();
    // ---- B: logits over 19 objects, softmax, * zo * 10 ----
    if (tid < 19*B_) {
      int n = tid / B_, b = tid - n*B_;
      const float* ipr = IP + (size_t)((b*T_ + t)*NB_ + 1 + n) * H_;
      const float* hwb = sc_hw + b*H_;
      float acc = 0.f;
      for (int k = 0; k < H_; ++k) acc += tanhf(hwb[k] + ipr[k]) * att_w[k];
      sh_al[n*B_ + b] = acc;
    }
    __syncthreads();
    if (tid < B_) {
      int b = tid;
      float m = -1e30f;
      for (int n = 0; n < 19; ++n) m = fmaxf(m, sh_al[n*B_+b]);
      float ex[19]; float s = 0.f;
      for (int n = 0; n < 19; ++n) { ex[n] = expf(sh_al[n*B_+b] - m); s += ex[n]; }
      float inv = 1.f / s;
      for (int n = 0; n < 19; ++n) {
        float zot = flags[(size_t)(b*T_ + t)*NB_ + 1 + n];
        sh_al[n*B_+b] = ex[n] * inv * zot * 10.f;
      }
    }
    __syncthreads();
    // ---- C: xin -> abuf ; ring slot gets entry h (f32 global + tanh bf16 LDS) ----
    for (int p = tid; p < B_*1024; p += NT) {
      int b = p >> 10, c = p & 1023;
      float v;
      if (c < H_) {
        const bf16* base = XEbf + (size_t)((b*T_ + t)*NB_ + 1) * H_ + c;
        float acc = 0.f;
        for (int n = 0; n < 19; ++n) acc += sh_al[n*B_+b] * (float)base[(size_t)n*H_];
        v = acc;
      } else {
        v = (float)XEbf[(size_t)((b*T_ + t)*NB_) * H_ + (c - H_)];
      }
      abuf[b*1536 + c] = (bf16)v;
    }
    for (int p = tid; p < B_*H_; p += NT) {
      int b = p >> 9, c = p & 511;
      float hv = sh_h[p];
      bufG[(size_t)(slot_new*B_ + b)*H_ + c] = hv;
      tbuf[(slot_new*B_ + b)*512 + c] = (bf16)tanhf(hv);
    }
    __syncthreads();
    // ---- D: frame attention  (WMMA over tanh ring, 4 N-chunks of 128) ----
    if (t >= 2) {
      for (int ch = 0; ch < 4; ++ch) {
        const int c0 = ch * 128;
        for (int idx = wv; idx < 56; idx += 16) {
          int mt2 = idx >> 3, nt2 = idx & 7;
          fx8 acc = gemm_tile(tbuf + mt2 * 16 * 512, 512,
                              fwT + (size_t)(c0 + nt2 * 16) * 512, 512, 512);
          store_acc_f32(acc, sc_lg, 128, mt2 * 16, nt2 * 16);
        }
        __syncthreads();
        for (int p = tid; p < B_*128; p += NT) {
          int b = p >> 7, cc = p & 127;
          float lgv[WIN_]; float mx = -1e30f;
#pragma unroll
          for (int w = 0; w < WIN_; ++w) {
            float v = -1e9f;
            if (w <= t) { int slot = (t - w) % WIN_; v = sc_lg[(slot*B_ + b)*128 + cc]; }
            lgv[w] = v; mx = fmaxf(mx, v);
          }
          float s = 0.f;
#pragma unroll
          for (int w = 0; w < WIN_; ++w) { lgv[w] = expf(lgv[w] - mx); s += lgv[w]; }
          float inv = 1.f / s;
          float hn = 0.f;
#pragma unroll
          for (int w = 0; w < WIN_; ++w) {
            if (w <= t) {
              int slot = (t - w) % WIN_;
              hn += bufG[(size_t)(slot*B_ + b)*H_ + c0 + cc] * lgv[w] * inv;
            }
          }
          sh_h[b*H_ + c0 + cc] = hn;
          abuf[b*1536 + 1024 + c0 + cc] = (bf16)hn;
        }
        __syncthreads();
      }
    }
    __syncthreads();
    // ---- E: GRU  (WMMA: Gsum = [xin|h] @ [wih;whh]^T ; Hg = h @ whh_g^T) ----
    for (int nt = wv; nt < 96; nt += 16) {
      fx8 acc = gemm_tile(abuf, 1536, wcatT + (size_t)nt * 16 * 1536, 1536, 1536);
      store_acc_bf16(acc, sc_gsum, 1536, 0, nt * 16);
    }
    for (int nt = wv; nt < 32; nt += 16) {
      fx8 acc = gemm_tile(abuf + 1024, 1536, whhg + (size_t)nt * 16 * 512, 512, 512);
      store_acc_bf16(acc, sc_hg, 512, 0, nt * 16);
    }
    __syncthreads();
    for (int p = tid; p < B_*H_; p += NT) {
      int b = p >> 9, j = p & 511;
      float gr  = (float)sc_gsum[b*1536 + j];
      float gz  = (float)sc_gsum[b*1536 + 512 + j];
      float gg  = (float)sc_gsum[b*1536 + 1024 + j];
      float hgr = (float)sc_hg[b*512 + j];
      float r = 1.f/(1.f + expf(-(gr + bih[j] + bhh[j])));
      float z = 1.f/(1.f + expf(-(gz + bih[512+j] + bhh[512+j])));
      float g = tanhf((gg - hgr + bih[1024+j]) + r * (hgr + bhh[1024+j]));
      float hn = (1.f - z)*g + z*sh_h[p];
      sh_h[p] = hn;
      abuf[b*1536 + 1024 + j] = (bf16)hn;
    }
    __syncthreads();
    // ---- F: classifier head + per-step loss + stream outs/hids ----
    for (int p = tid; p < B_*64; p += NT) {
      int b = p >> 6, j = p & 63;
      const float* hb = sh_h + b*H_;
      float acc = d1b[j];
      for (int k = 0; k < H_; ++k) acc += hb[k]*d1w[k*64 + j];
      sh_t64[p] = fmaxf(acc, 0.f);
    }
    __syncthreads();
    if (tid < 20) {
      int b = tid >> 1, c = tid & 1;
      float acc = d2b[c];
      for (int k = 0; k < 64; ++k) acc += sh_t64[b*64+k]*d2w[k*2+c];
      sh_o[tid] = acc;
      out[2 + (t*B_ + b)*2 + c] = acc;
    }
    __syncthreads();
    if (tid == 0) {
      float lt = 0.f;
      for (int b = 0; b < B_; ++b) {
        float o0 = sh_o[b*2], o1 = sh_o[b*2+1];
        float m = fmaxf(o0, o1);
        float lse = m + logf(expf(o0-m)+expf(o1-m));
        int cls = (int)(y[b*2+1] + 0.5f);
        float ce = lse - (cls ? o1 : o0);
        float pen = -fmaxf(0.f, (toa[b] - (float)t - 1.f) * (1.f/20.f));
        lt += expf(pen)*ce*y[b*2+1] + ce*y[b*2+0];
      }
      sh_ce[0] += lt * (1.f/(float)B_);
    }
    for (int p = tid; p < B_*H_; p += NT)
      out[2 + T_*B_*2 + (size_t)t*B_*H_ + p] = sh_h[p];
    __syncthreads();
  }
  if (tid == 0) out[0] = sh_ce[0];
}

// ---------------- Post-scan self-aggregation attention -------------------------------
__global__ void attn_kernel(const float* __restrict__ hids, float* __restrict__ attn)
{
  __shared__ float e[T_];
  int b = blockIdx.x / T_;
  int t = blockIdx.x - b*T_;
  int s = threadIdx.x;
  if (s < T_) {
    const float* ht = hids + (size_t)(t*B_ + b)*H_;
    const float* hs = hids + (size_t)(s*B_ + b)*H_;
    float acc = 0.f;
    for (int k = 0; k < H_; ++k) acc += ht[k]*hs[k];
    e[s] = 2.f*acc;
  }
  __syncthreads();
  if (threadIdx.x == 0) {
    float m = -1e30f;
    for (int i = 0; i < T_; ++i) m = fmaxf(m, e[i]);
    float sum = 0.f;
    for (int i = 0; i < T_; ++i) { e[i] = expf(e[i]-m); sum += e[i]; }
    float inv = 1.f/sum;
    for (int i = 0; i < T_; ++i) attn[(size_t)blockIdx.x*T_ + i] = e[i]*inv;
  }
}

__global__ void feat_kernel(const float* __restrict__ attn, const float* __restrict__ hids,
                            const float* __restrict__ saa, float* __restrict__ feat)
{
  __shared__ float w2[T_];
  int b = blockIdx.x;
  if (threadIdx.x < T_) {
    int s = threadIdx.x;
    float acc = 0.f;
    for (int t = 0; t < T_; ++t) acc += saa[t]*attn[((size_t)b*T_ + t)*T_ + s];
    w2[s] = acc;
  }
  __syncthreads();
  for (int c = threadIdx.x; c < 1024; c += 256) {
    int hcol = c & 511;
    float acc = 0.f;
    for (int s = 0; s < T_; ++s) acc += w2[s]*hids[(size_t)(s*B_ + b)*H_ + hcol];
    feat[b*1024 + c] = acc;
  }
}

__global__ void aux_kernel(const float* __restrict__ feat,
                           const float* __restrict__ a1w, const float* __restrict__ a1b,
                           const float* __restrict__ a2w, const float* __restrict__ a2b,
                           const float* __restrict__ y, float* __restrict__ out)
{
  __shared__ float t64[640];
  __shared__ float dec[20];
  for (int p = threadIdx.x; p < 640; p += 64) {
    int b = p >> 6, j = p & 63;
    float acc = a1b[j];
    for (int c = 0; c < 1024; ++c) acc += feat[b*1024 + c]*a1w[c*64 + j];
    t64[p] = fmaxf(acc, 0.f);
  }
  __syncthreads();
  if (threadIdx.x < 20) {
    int b = threadIdx.x >> 1, cc = threadIdx.x & 1;
    float acc = a2b[cc];
    for (int k = 0; k < 64; ++k) acc += t64[b*64+k]*a2w[k*2+cc];
    dec[threadIdx.x] = acc;
  }
  __syncthreads();
  if (threadIdx.x == 0) {
    float s = 0.f;
    for (int b = 0; b < B_; ++b) {
      float o0 = dec[b*2], o1 = dec[b*2+1];
      float m = fmaxf(o0, o1);
      float lse = m + logf(expf(o0-m)+expf(o1-m));
      int cls = (int)(y[b*2+1] + 0.5f);
      s += lse - (cls ? o1 : o0);
    }
    out[1] = s * (1.f/(float)B_);
  }
}

// ---------------- host launcher ------------------------------------------------------
extern "C" void kernel_launch(void* const* d_in, const int* in_sizes, int n_in,
                              void* d_out, int out_size, void* d_ws, size_t ws_size,
                              hipStream_t stream)
{
  (void)in_sizes; (void)n_in; (void)out_size; (void)ws_size;
  const float* x       = (const float*)d_in[0];
  const float* y       = (const float*)d_in[1];
  const float* toa     = (const float*)d_in[2];
  const float* phi_w   = (const float*)d_in[4];
  const float* phi_b   = (const float*)d_in[5];
  const float* att_w   = (const float*)d_in[6];
  const float* att_ua  = (const float*)d_in[7];
  const float* att_ba  = (const float*)d_in[8];
  const float* att_wa  = (const float*)d_in[9];
  const float* frame_w = (const float*)d_in[10];
  const float* wih     = (const float*)d_in[11];
  const float* whh     = (const float*)d_in[12];
  const float* bih     = (const float*)d_in[13];
  const float* bhh     = (const float*)d_in[14];
  const float* d1w     = (const float*)d_in[15];
  const float* d1b     = (const float*)d_in[16];
  const float* d2w     = (const float*)d_in[17];
  const float* d2b     = (const float*)d_in[18];
  const float* saa     = (const float*)d_in[19];
  const float* a1w     = (const float*)d_in[20];
  const float* a1b     = (const float*)d_in[21];
  const float* a2w     = (const float*)d_in[22];
  const float* a2b     = (const float*)d_in[23];

  char* ws = (char*)d_ws;
  size_t off = 0;
  auto carve = [&](size_t bytes) { char* p = ws + off; off += (bytes + 255) & ~(size_t)255; return p; };
  bf16*  xbf   = (bf16*) carve((size_t)NROWS_PAD * D_ * 2);
  float* flags = (float*)carve((size_t)NROWS * 4);
  bf16*  phiT  = (bf16*) carve((size_t)H_ * D_ * 2);
  bf16*  uaT   = (bf16*) carve((size_t)H_ * H_ * 2);
  bf16*  awaT  = (bf16*) carve((size_t)H_ * H_ * 2);
  bf16*  fwT   = (bf16*) carve((size_t)H_ * H_ * 2);
  bf16*  wcatT = (bf16*) carve((size_t)1536 * 1536 * 2);
  bf16*  whhg  = (bf16*) carve((size_t)H_ * H_ * 2);
  bf16*  xebf  = (bf16*) carve((size_t)NROWS_PAD * H_ * 2);
  float* ip    = (float*)carve((size_t)NROWS * H_ * 4);
  float* bufG  = (float*)carve((size_t)WIN_ * B_ * H_ * 4);
  float* attn  = (float*)carve((size_t)B_ * T_ * T_ * 4);
  float* feat  = (float*)carve((size_t)B_ * 1024 * 4);

  float* out = (float*)d_out;

  convert_zo<<<NROWS, 256, 0, stream>>>(x, xbf, flags);
  transpose_bf16<<<(D_*H_ + 255)/256, 256, 0, stream>>>(phi_w, phiT, D_, H_);
  transpose_bf16<<<(H_*H_ + 255)/256, 256, 0, stream>>>(att_ua, uaT, H_, H_);
  transpose_bf16<<<(H_*H_ + 255)/256, 256, 0, stream>>>(att_wa, awaT, H_, H_);
  transpose_bf16<<<(H_*H_ + 255)/256, 256, 0, stream>>>(frame_w, fwT, H_, H_);
  build_wcat<<<(1536*1536 + 255)/256, 256, 0, stream>>>(wih, whh, wcatT);
  cast_bf16<<<(H_*H_ + 255)/256, 256, 0, stream>>>(whh + 1024*512, whhg, H_*H_);

  wmma_gemm_n512<<<MTILES * 4, 256, 0, stream>>>(xbf, phiT, phi_b, nullptr, xebf, D_, 1, NROWS);
  wmma_gemm_n512<<<MTILES * 4, 256, 0, stream>>>(xebf, uaT, att_ba, ip, nullptr, H_, 0, NROWS);

  scan_kernel<<<1, 512, SCAN_SMEM, stream>>>(xebf, ip, flags, awaT, att_w, fwT, wcatT, whhg,
                                             bih, bhh, d1w, d1b, d2w, d2b,
                                             y, toa, bufG, out);

  attn_kernel<<<B_*T_, 128, 0, stream>>>(out + 2002, attn);
  feat_kernel<<<B_, 256, 0, stream>>>(attn, out + 2002, saa, feat);
  aux_kernel<<<1, 64, 0, stream>>>(feat, a1w, a1b, a2w, a2b, y, out);
}
